// MultiParticlesGraphNet_30760555774211
// MI455X (gfx1250) — matrix operand order repaired and checked by reading, not compiled, not run on hardware
//
#include <hip/hip_runtime.h>
#include <hip/hip_bf16.h>

#define H 128
#define NLAYERS 4

typedef __bf16 bf16_t;
typedef __attribute__((ext_vector_type(8)))  __bf16 v8bf;
typedef __attribute__((ext_vector_type(16))) __bf16 v16bf;
typedef __attribute__((ext_vector_type(8)))  float  v8f;
typedef int v4i __attribute__((vector_size(16)));

#if defined(__HIP_DEVICE_COMPILE__) && __has_builtin(__builtin_amdgcn_global_load_async_to_lds_b128)
#define HAVE_ASYNC_LDS 1
#endif

__device__ __forceinline__ float gelu_erf(float x) {
  return 0.5f * x * (1.0f + erff(x * 0.70710678118654752f));
}

// ---------------- async LDS staging (CDNA5 GLOBAL_LOAD_ASYNC_TO_LDS, ASYNCcnt) ----------------
__device__ __forceinline__ void copy16_to_lds(void* ldsp, const void* gp) {
#ifdef HAVE_ASYNC_LDS
  typedef __attribute__((address_space(1))) v4i gv4i;
  typedef __attribute__((address_space(3))) v4i lv4i;
  __builtin_amdgcn_global_load_async_to_lds_b128((gv4i*)gp, (lv4i*)ldsp, 0, 0);
#else
  *(float4*)ldsp = *(const float4*)gp;
#endif
}

__device__ __forceinline__ void wait_async_lds() {
#ifdef HAVE_ASYNC_LDS
#if __has_builtin(__builtin_amdgcn_s_wait_asynccnt)
  __builtin_amdgcn_s_wait_asynccnt(0);
#else
  asm volatile("s_wait_asynccnt 0x0" ::: "memory");
#endif
#endif
}

// Contiguous bf16 copy global -> LDS, 16B granules, whole block (nthr threads).
__device__ __forceinline__ void stage_bytes(void* lds, const void* g, int bytes, int tid, int nthr) {
  for (int ofs = tid * 16; ofs < bytes; ofs += nthr * 16)
    copy16_to_lds((char*)lds + ofs, (const char*)g + ofs);
}

// ---- WMMA fragment loaders (bf16 16x16x32, wave32 layouts per CDNA5 ISA) ----
__device__ __forceinline__ v16bf frag_a(const bf16_t* base, int lda, int m0, int k0, int lane) {
  int row = m0 + (lane & 15);
  int kb  = k0 + ((lane >> 4) << 3);
  const bf16_t* p = base + row * lda + kb;
  v8bf lo = *(const v8bf*)(p);
  v8bf hi = *(const v8bf*)(p + 16);
  return __builtin_shufflevector(lo, hi, 0,1,2,3,4,5,6,7,8,9,10,11,12,13,14,15);
}

// B from LDS weights stored TRANSPOSED: Wt[n][k] (ldk = K dim).
__device__ __forceinline__ v16bf frag_b(const bf16_t* wt, int ldk, int n0, int k0, int lane) {
  int n  = n0 + (lane & 15);
  int kb = k0 + ((lane >> 4) << 3);
  const bf16_t* p = wt + n * ldk + kb;
  v8bf lo = *(const v8bf*)(p);
  v8bf hi = *(const v8bf*)(p + 16);
  return __builtin_shufflevector(lo, hi, 0,1,2,3,4,5,6,7,8,9,10,11,12,13,14,15);
}

// One wave: 16x128 tile of (A[M x kdim] @ W[kdim x 128]) into acc[8].
__device__ __forceinline__ void gemm_rowtile(const bf16_t* A, int lda, const bf16_t* Wt,
                                             int kdim, int m0, int lane, v8f acc[8]) {
  for (int k0 = 0; k0 < kdim; k0 += 32) {
    v16bf a = frag_a(A, lda, m0, k0, lane);
#pragma unroll
    for (int nt = 0; nt < 8; ++nt) {
      v16bf b = frag_b(Wt, kdim, nt * 16, k0, lane);
      acc[nt] = __builtin_amdgcn_wmma_f32_16x16x32_bf16(
          false, a, false, b, (short)0, acc[nt], false, false);
    }
  }
}

// ------------- one-shot weight prep: f32 row-major [kdim][128] -> bf16 transposed [128][kdim] -------------
__global__ void __launch_bounds__(256)
wprep_kernel(const float* __restrict__ W, bf16_t* __restrict__ Wt, int kdim) {
  int idx = blockIdx.x * 256 + threadIdx.x;
  if (idx < kdim * H) {
    int k = idx >> 7;
    int n = idx & 127;
    Wt[n * kdim + k] = (bf16_t)W[idx];
  }
}

// ------------------------------ LayerNorm ------------------------------
__global__ void __launch_bounds__(128)
ln_kernel(const float* __restrict__ h, const float* __restrict__ g,
          const float* __restrict__ bta, float* __restrict__ hn,
          bf16_t* __restrict__ hnb) {
  __shared__ float red[128];
  int row = blockIdx.x, tid = threadIdx.x;
  float x = h[row * H + tid];
  red[tid] = x;
  __syncthreads();
  for (int s = 64; s > 0; s >>= 1) { if (tid < s) red[tid] += red[tid + s]; __syncthreads(); }
  float mu = red[0] * (1.0f / H);
  __syncthreads();
  float d = x - mu;
  red[tid] = d * d;
  __syncthreads();
  for (int s = 64; s > 0; s >>= 1) { if (tid < s) red[tid] += red[tid + s]; __syncthreads(); }
  float var = red[0] * (1.0f / H);
  float y = d * rsqrtf(var + 1e-5f) * g[tid] + bta[tid];
  hn[row * H + tid] = y;
  hnb[row * H + tid] = (bf16_t)y;
}

// ------------------------------ Edge pipeline ------------------------------
// One WG (256 thr / 8 waves) per pair of nodes (b, i0), (b, i0+1): coord-MLP,
// edge-MLP (2 WMMA GEMMs) and segment-sum (contiguous 63-edge block, no atomics).
// Waves 0-3 own node 0's four 16-row M-tiles; waves 4-7 own node 1's.
__global__ void __launch_bounds__(256)
edge_kernel(const bf16_t* __restrict__ hnb, const float* __restrict__ coord,
            const float* __restrict__ cW1, const float* __restrict__ cb1,
            const bf16_t* __restrict__ CW2Tg, const float* __restrict__ cb2,
            const bf16_t* __restrict__ EW1Tg, const float* __restrict__ eb1,
            const bf16_t* __restrict__ EW2Tg, const float* __restrict__ eb2,
            float* __restrict__ agg) {
  extern __shared__ float4 smem_raw[];
  char* smem   = (char*)smem_raw;
  bf16_t* EW1T = (bf16_t*)smem;            // 128*384  = 96 KB
  bf16_t* EW2T = EW1T + 128 * 384;         // 128*128  = 32 KB
  bf16_t* CW2T = EW2T + 128 * 128;         // 128*128  = 32 KB
  bf16_t* HN   = CW2T + 128 * 128;         // 64*128   = 16 KB
  bf16_t* A0   = HN   + 64 * 128;          // 64*384   = 48 KB
  bf16_t* A1   = A0   + 64 * 384;          // 64*384   = 48 KB
  bf16_t* MID0 = A1   + 64 * 384;          // 64*128   = 16 KB
  bf16_t* MID1 = MID0 + 64 * 128;          // 64*128   = 16 KB   (total 304 KB)
  float*  OUTF0 = (float*)A0;              // aliases (f32 64*128 fits)
  float*  OUTF1 = (float*)A1;

  int blk = blockIdx.x;                    // 1024 blocks
  int b   = blk >> 5;
  int i0  = (blk & 31) * 2;
  int tid = threadIdx.x, lane = tid & 31, wave = tid >> 5;
  int nsel = wave >> 2;                    // which node this wave works on
  int m0   = (wave & 3) * 16;
  int isel = i0 + nsel;
  bf16_t* Asel   = nsel ? A1 : A0;
  bf16_t* MIDsel = nsel ? MID1 : MID0;
  float*  OUTFsel = nsel ? OUTF1 : OUTF0;

  // Async-DMA stage: pre-converted bf16 weights + this graph's hn block.
  stage_bytes(EW1T, EW1Tg, 128 * 384 * 2, tid, 256);
  stage_bytes(EW2T, EW2Tg, 128 * 128 * 2, tid, 256);
  stage_bytes(CW2T, CW2Tg, 128 * 128 * 2, tid, 256);
  stage_bytes(HN, hnb + (size_t)(b * 64) * H, 64 * 128 * 2, tid, 256);
  wait_async_lds();
  __syncthreads();

  // cd1 = gelu(dx*cW1[0,:] + dy*cW1[1,:] + cb1)  (2 nodes x 64 targets; j==i is padding)
  float c0x = coord[(b * 64 + i0) * 2],     c0y = coord[(b * 64 + i0) * 2 + 1];
  float c1x = coord[(b * 64 + i0 + 1) * 2], c1y = coord[(b * 64 + i0 + 1) * 2 + 1];
  for (int t = tid; t < 2 * 64 * 128; t += 256) {
    int n = t >> 13, rem = t & 8191;
    int j = rem >> 7, c = rem & 127;
    float sx = n ? c1x : c0x, sy = n ? c1y : c0y;
    float dx = sx - coord[(b * 64 + j) * 2];
    float dy = sy - coord[(b * 64 + j) * 2 + 1];
    float v = dx * cW1[c] + dy * cW1[H + c] + cb1[c];
    (n ? MID1 : MID0)[rem] = (bf16_t)gelu_erf(v);
  }
  __syncthreads();

  // cd = gelu(cd1 @ cW2 + cb2) -> Asel[:, 256:384]
  {
    v8f acc[8] = {};
    gemm_rowtile(MIDsel, 128, CW2T, 128, m0, lane, acc);
#pragma unroll
    for (int nt = 0; nt < 8; ++nt) {
      int col = nt * 16 + (lane & 15);
      float bias = cb2[col];
#pragma unroll
      for (int r = 0; r < 8; ++r) {
        int row = m0 + r + ((lane >> 4) << 3);
        Asel[row * 384 + 256 + col] = (bf16_t)gelu_erf(acc[nt][r] + bias);
      }
    }
  }
  // A[:, 0:128] = hn[i] broadcast, A[:, 128:256] = hn[j]
  for (int t = tid; t < 2 * 64 * 256; t += 256) {
    int n = t >> 14, rem = t & 16383;
    int j = rem >> 8, c = rem & 255;
    int ii = i0 + n;
    (n ? A1 : A0)[j * 384 + c] = (c < 128) ? HN[ii * 128 + c] : HN[j * 128 + (c - 128)];
  }
  __syncthreads();

  // ef = gelu(A @ eW1 + eb1) -> MIDsel
  {
    v8f acc[8] = {};
    gemm_rowtile(Asel, 384, EW1T, 384, m0, lane, acc);
#pragma unroll
    for (int nt = 0; nt < 8; ++nt) {
      int col = nt * 16 + (lane & 15);
      float bias = eb1[col];
#pragma unroll
      for (int r = 0; r < 8; ++r) {
        int row = m0 + r + ((lane >> 4) << 3);
        MIDsel[row * 128 + col] = (bf16_t)gelu_erf(acc[nt][r] + bias);
      }
    }
  }
  __syncthreads();  // all A reads done before OUTF (alias of A) is written

  // ef = gelu(MID @ eW2 + eb2); zero padded self-edge row; -> OUTF (f32)
  {
    v8f acc[8] = {};
    gemm_rowtile(MIDsel, 128, EW2T, 128, m0, lane, acc);
#pragma unroll
    for (int nt = 0; nt < 8; ++nt) {
      int col = nt * 16 + (lane & 15);
      float bias = eb2[col];
#pragma unroll
      for (int r = 0; r < 8; ++r) {
        int row = m0 + r + ((lane >> 4) << 3);
        float v = gelu_erf(acc[nt][r] + bias);
        OUTFsel[row * 128 + col] = (row == isel) ? 0.0f : v;
      }
    }
  }
  __syncthreads();

  // segment-sum: threads 0..127 -> node i0, 128..255 -> node i0+1
  {
    int n = tid >> 7, c = tid & 127;
    const float* of = n ? OUTF1 : OUTF0;
    float s = 0.0f;
    for (int j = 0; j < 64; ++j) s += of[j * 128 + c];
    agg[(size_t)(b * 64 + i0 + n) * H + c] = s;
  }
}

// ------------------------------ Node MLP + residual ------------------------------
__global__ void __launch_bounds__(256)
node_kernel(const float* __restrict__ hn, const bf16_t* __restrict__ hnb,
            const float* __restrict__ agg,
            const bf16_t* __restrict__ NW1Tg, const float* __restrict__ nb1,
            const bf16_t* __restrict__ NW2Tg, const float* __restrict__ nb2,
            float* __restrict__ hout) {
  extern __shared__ float4 smem_raw[];
  char* smem  = (char*)smem_raw;
  bf16_t* W1T = (bf16_t*)smem;           // 128*256 = 64 KB
  bf16_t* W2T = W1T + 128 * 256;         // 128*128 = 32 KB
  bf16_t* A   = W2T + 128 * 128;         // 128*256 = 64 KB
  bf16_t* MID = A   + 128 * 256;         // 128*128 = 32 KB  (total 192 KB)

  int r0 = blockIdx.x * 128;             // 16 blocks x 128 rows
  int tid = threadIdx.x, lane = tid & 31, m0 = (tid >> 5) * 16;

  stage_bytes(W1T, NW1Tg, 128 * 256 * 2, tid, 256);
  stage_bytes(W2T, NW2Tg, 128 * 128 * 2, tid, 256);
  wait_async_lds();
  for (int t = tid; t < 128 * 256; t += 256) {
    int j = t >> 8, c = t & 255;
    A[j * 256 + c] = (c < 128) ? hnb[(size_t)(r0 + j) * H + c]
                               : (bf16_t)agg[(size_t)(r0 + j) * H + (c - 128)];
  }
  __syncthreads();

  // gelu([hn|agg] @ nW1 + nb1) -> MID
  {
    v8f acc[8] = {};
    gemm_rowtile(A, 256, W1T, 256, m0, lane, acc);
#pragma unroll
    for (int nt = 0; nt < 8; ++nt) {
      int col = nt * 16 + (lane & 15);
      float bias = nb1[col];
#pragma unroll
      for (int r = 0; r < 8; ++r) {
        int row = m0 + r + ((lane >> 4) << 3);
        MID[row * 128 + col] = (bf16_t)gelu_erf(acc[nt][r] + bias);
      }
    }
  }
  __syncthreads();

  // h = hn + (MID @ nW2 + nb2)
  {
    v8f acc[8] = {};
    gemm_rowtile(MID, 128, W2T, 128, m0, lane, acc);
#pragma unroll
    for (int nt = 0; nt < 8; ++nt) {
      int col = nt * 16 + (lane & 15);
      float bias = nb2[col];
#pragma unroll
      for (int r = 0; r < 8; ++r) {
        int row = m0 + r + ((lane >> 4) << 3);
        size_t gi = (size_t)(r0 + row) * H + col;
        hout[gi] = hn[gi] + acc[nt][r] + bias;
      }
    }
  }
}

// ------------------------------ Host orchestration ------------------------------
extern "C" void kernel_launch(void* const* d_in, const int* in_sizes, int n_in,
                              void* d_out, int out_size, void* d_ws, size_t ws_size,
                              hipStream_t stream) {
  const float* h0    = (const float*)d_in[0];
  const float* coord = (const float*)d_in[1];
  const float* ln_g  = (const float*)d_in[2];
  const float* ln_b  = (const float*)d_in[3];
  const float* cW1   = (const float*)d_in[4];
  const float* cb1   = (const float*)d_in[5];
  const float* cW2   = (const float*)d_in[6];
  const float* cb2   = (const float*)d_in[7];
  const float* eW1   = (const float*)d_in[8];
  const float* eb1   = (const float*)d_in[9];
  const float* eW2   = (const float*)d_in[10];
  const float* eb2   = (const float*)d_in[11];
  const float* nW1   = (const float*)d_in[12];
  const float* nb1   = (const float*)d_in[13];
  const float* nW2   = (const float*)d_in[14];
  const float* nb2   = (const float*)d_in[15];

  char* ws = (char*)d_ws;
  float*  hn_f = (float*)(ws);                        // 2048*128 f32 = 1 MB
  float*  aggb = (float*)(ws + (1u << 20));           // 1 MB
  float*  hbuf = (float*)(ws + (2u << 20));           // 1 MB
  bf16_t* hn_b = (bf16_t*)(ws + (3u << 20));          // 0.5 MB
  bf16_t* wpool = (bf16_t*)(ws + 3670016u);           // 4 layers x 256 KB bf16 weights

  // per-layer offsets (bf16 elems) inside wpool
  const int L_CW2T = 0;                 // 128*128
  const int L_EW1T = 16384;             // 128*384
  const int L_EW2T = 65536;             // 128*128
  const int L_NW1T = 81920;             // 128*256
  const int L_NW2T = 114688;            // 128*128
  const int L_STRIDE = 131072;

  // One-shot f32 -> transposed bf16 weight prep (runs every call; deterministic).
  for (int l = 0; l < NLAYERS; ++l) {
    bf16_t* lp = wpool + l * L_STRIDE;
    wprep_kernel<<<64,  256, 0, stream>>>(cW2 + l * H * H,     lp + L_CW2T, 128);
    wprep_kernel<<<192, 256, 0, stream>>>(eW1 + l * 3 * H * H, lp + L_EW1T, 384);
    wprep_kernel<<<64,  256, 0, stream>>>(eW2 + l * H * H,     lp + L_EW2T, 128);
    wprep_kernel<<<128, 256, 0, stream>>>(nW1 + l * 2 * H * H, lp + L_NW1T, 256);
    wprep_kernel<<<64,  256, 0, stream>>>(nW2 + l * H * H,     lp + L_NW2T, 128);
  }

  const size_t edge_lds = (size_t)(128 * 384 + 2 * 128 * 128 + 64 * 128 +
                                   2 * 64 * 384 + 2 * 64 * 128) * sizeof(bf16_t);  // 304 KB
  const size_t node_lds = (size_t)(128 * 256 + 128 * 128 + 128 * 256 + 128 * 128) * sizeof(bf16_t);

  for (int l = 0; l < NLAYERS; ++l) {
    const float* hin = (l == 0) ? h0 : hbuf;
    float* ho = (l == NLAYERS - 1) ? (float*)d_out : hbuf;
    bf16_t* lp = wpool + l * L_STRIDE;

    ln_kernel<<<2048, 128, 0, stream>>>(hin, ln_g + l * H, ln_b + l * H, hn_f, hn_b);

    edge_kernel<<<1024, 256, edge_lds, stream>>>(
        hn_b, coord,
        cW1 + l * 2 * H, cb1 + l * H,
        lp + L_CW2T, cb2 + l * H,
        lp + L_EW1T, eb1 + l * H,
        lp + L_EW2T, eb2 + l * H,
        aggb);

    node_kernel<<<16, 256, node_lds, stream>>>(
        hn_f, hn_b, aggb,
        lp + L_NW1T, nb1 + l * H,
        lp + L_NW2T, nb2 + l * H,
        ho);
  }
}
